// classifier_824633721327
// MI455X (gfx1250) — matrix operand order repaired and checked by reading, not compiled, not run on hardware
//
#include <hip/hip_runtime.h>
#include <hip/hip_bf16.h>
#include <math.h>

typedef __attribute__((ext_vector_type(16))) __bf16 v16bf;
typedef __attribute__((ext_vector_type(8)))  float  v8f;

#define WMMA_BF16(a, b, c) \
  __builtin_amdgcn_wmma_f32_16x16x32_bf16(false, (a), false, (b), (short)0, (c), false, false)

__device__ __forceinline__ float gelu_exact(float x) {
  return 0.5f * x * (1.0f + erff(x * 0.70710678118654752f));
}

// ---------------------------------------------------------------------------
// Kernel 1: mask trilinear 64^3 -> 8^3 (collapses to 8-corner mean) fused with
// tok[b,s,d] = x[b,d,s] * m8[b,s] + pe[d,s]
// ---------------------------------------------------------------------------
__global__ __launch_bounds__(256) void tok_init_kernel(
    const float* __restrict__ x, const float* __restrict__ mask,
    const float* __restrict__ pe, float* __restrict__ tok) {
  int bid = blockIdx.x;             // 0..2047
  int b = bid >> 9;
  int s = bid & 511;
  __shared__ float m8s;
  if (threadIdx.x == 0) {
    int a = s >> 6, bb = (s >> 3) & 7, c = s & 7;
    int ia = a * 8 + 3, ib = bb * 8 + 3, ic = c * 8 + 3;
    const float* mp = mask + (size_t)b * 262144;  // 64*64*64
    float sum = 0.0f;
#pragma unroll
    for (int dz = 0; dz < 2; ++dz)
#pragma unroll
      for (int dy = 0; dy < 2; ++dy)
#pragma unroll
        for (int dx = 0; dx < 2; ++dx)
          sum += mp[(size_t)(ia + dz) * 4096 + (ib + dy) * 64 + (ic + dx)];
    m8s = sum * 0.125f;
  }
  __syncthreads();
  float m8 = m8s;
  int d = threadIdx.x;
#pragma unroll
  for (int i = 0; i < 2; ++i, d += 256) {
    tok[((size_t)(b * 512 + s)) * 512 + d] =
        x[((size_t)(b * 512 + d)) * 512 + s] * m8 + pe[(size_t)d * 512 + s];
  }
}

// ---------------------------------------------------------------------------
// Kernel 2: WMMA bf16 GEMM  C[M,N] = A[M,K] @ B[K,N] + bias (+GELU)
// Block tile 128x64, 8 waves stacked on M, each wave owns 16x64 (4 accums).
// Register-staged double-buffered LDS pipeline: prefetch next f32 tile into
// VGPRs while WMMAs consume current bf16 LDS buffer. One barrier per K-step.
// epi: 0 = bias only, 1 = bias + exact GELU
// ---------------------------------------------------------------------------
__global__ __launch_bounds__(256) void gemm_bf16_kernel(
    const float* __restrict__ A, const float* __restrict__ B,
    const float* __restrict__ bias, float* __restrict__ C,
    int M, int N, int K, int epi) {
  __shared__ __bf16 As[2][128][40];  // [buf][m][k]
  __shared__ __bf16 Bs[2][64][40];   // [buf][n][k] (transposed: contiguous frags)
  int tid = threadIdx.x;
  int bm = blockIdx.y * 128, bn = blockIdx.x * 64;
  int wv = tid >> 5, lane = tid & 31;
  int wm = wv * 16;                  // wave's M offset within tile
  int lrow = lane & 15, lhi = lane >> 4;

  int ar = tid >> 1, ac = (tid & 1) * 16;  // A tile: 128 rows x 32 k, 16/thread
  int bk = tid >> 3, bnn = (tid & 7) * 8;  // B tile: 32 k x 64 n, 8/thread

  v8f acc0 = {}, acc1 = {}, acc2 = {}, acc3 = {};
  float aR[16];
  float bR[8];

  {  // prologue: tile 0 -> regs -> LDS buf 0
    const float* Ap = A + (size_t)(bm + ar) * K + ac;
#pragma unroll
    for (int i = 0; i < 16; ++i) aR[i] = Ap[i];
    const float* Bp = B + (size_t)bk * N + bn + bnn;
#pragma unroll
    for (int i = 0; i < 8; ++i) bR[i] = Bp[i];
#pragma unroll
    for (int i = 0; i < 16; ++i) As[0][ar][ac + i] = (__bf16)aR[i];
#pragma unroll
    for (int i = 0; i < 8; ++i) Bs[0][bnn + i][bk] = (__bf16)bR[i];
  }

  int buf = 0;
  for (int k0 = 0; k0 < K; k0 += 32, buf ^= 1) {
    __syncthreads();  // LDS tile[buf] ready for all waves
    bool more = (k0 + 32) < K;
    if (more) {  // global prefetch of next tile into registers
      const float* Ap = A + (size_t)(bm + ar) * K + k0 + 32 + ac;
#pragma unroll
      for (int i = 0; i < 16; ++i) aR[i] = Ap[i];
      const float* Bp = B + (size_t)(k0 + 32 + bk) * N + bn + bnn;
#pragma unroll
      for (int i = 0; i < 8; ++i) bR[i] = Bp[i];
    }

    v16bf af, bf0, bf1, bf2, bf3;
#pragma unroll
    for (int j = 0; j < 8; ++j) {
      af[j]     = As[buf][wm + lrow][lhi * 16 + j];
      af[8 + j] = As[buf][wm + lrow][lhi * 16 + 8 + j];
    }
#pragma unroll
    for (int e = 0; e < 16; ++e) {
      bf0[e] = Bs[buf][lrow][lhi * 16 + e];
      bf1[e] = Bs[buf][16 + lrow][lhi * 16 + e];
      bf2[e] = Bs[buf][32 + lrow][lhi * 16 + e];
      bf3[e] = Bs[buf][48 + lrow][lhi * 16 + e];
    }
    acc0 = WMMA_BF16(af, bf0, acc0);
    acc1 = WMMA_BF16(af, bf1, acc1);
    acc2 = WMMA_BF16(af, bf2, acc2);
    acc3 = WMMA_BF16(af, bf3, acc3);

    if (more) {  // convert + store next tile into the other buffer
#pragma unroll
      for (int i = 0; i < 16; ++i) As[buf ^ 1][ar][ac + i] = (__bf16)aR[i];
#pragma unroll
      for (int i = 0; i < 8; ++i) Bs[buf ^ 1][bnn + i][bk] = (__bf16)bR[i];
    }
  }

  // epilogue: each wave writes its 16x64 slab
#pragma unroll
  for (int q = 0; q < 4; ++q) {
    v8f acc = (q == 0) ? acc0 : (q == 1) ? acc1 : (q == 2) ? acc2 : acc3;
    int col = bn + q * 16 + lrow;
    float bv = bias ? bias[col] : 0.0f;
#pragma unroll
    for (int r = 0; r < 8; ++r) {
      int row = bm + wm + r + lhi * 8;
      float v = acc[r] + bv;
      if (epi == 1) v = gelu_exact(v);
      C[(size_t)row * N + col] = v;
    }
  }
}

// ---------------------------------------------------------------------------
// Kernel 3: fused attention, one block per (b, h, 16-row q tile).
// scores via WMMA (K-frags straight from qkv), full-row softmax, P@V via WMMA
// with V staged transposed in LDS, gate-mix with memout on layer 0.
// ---------------------------------------------------------------------------
__global__ __launch_bounds__(256) void attn_kernel(
    const float* __restrict__ qkv, const float* __restrict__ memout,
    const float* __restrict__ gate, float* __restrict__ attnout, int layer0) {
  int bid = blockIdx.x;  // ((b*8 + h)*32 + st)
  int st = bid & 31, h = (bid >> 5) & 7, b = bid >> 8;
  int s0 = st * 16;
  __shared__ __bf16 qs[16][72];
  __shared__ float  S[16][520];
  __shared__ __bf16 Vt[64][72];       // [dim][token-in-chunk]
  __shared__ float  Oacc[2][16][66];
  __shared__ float  red[16][16];

  int tid = threadIdx.x, wv = tid >> 5, lane = tid & 31;
  int lrow = lane & 15, lhi = lane >> 4;
  const float scale = 0.125f;  // 1/sqrt(64)

  {  // stage q tile (bf16)
    int row = tid >> 4, f0 = (tid & 15) * 4;
    const float* qp = qkv + ((size_t)(b * 512 + s0 + row)) * 1536 + h * 64 + f0;
#pragma unroll
    for (int i = 0; i < 4; ++i) qs[row][f0 + i] = (__bf16)qp[i];
  }
  __syncthreads();

  // ---- scores: 16 x 512, 8 waves x 16 columns per sweep ----
  for (int c0 = 0; c0 < 512; c0 += 128) {
    int t = c0 + wv * 16 + lrow;  // key token owned by this lane (B column)
    const float* kp = qkv + ((size_t)(b * 512 + t)) * 1536 + 512 + h * 64;
    v8f acc = {};
#pragma unroll
    for (int ks = 0; ks < 2; ++ks) {
      v16bf af, bfr;
#pragma unroll
      for (int j = 0; j < 8; ++j) {
        af[j]     = qs[lrow][ks * 32 + lhi * 16 + j];
        af[8 + j] = qs[lrow][ks * 32 + lhi * 16 + 8 + j];
      }
#pragma unroll
      for (int e = 0; e < 16; ++e) bfr[e] = (__bf16)kp[ks * 32 + lhi * 16 + e];
      acc = WMMA_BF16(af, bfr, acc);
    }
#pragma unroll
    for (int r = 0; r < 8; ++r)
      S[r + lhi * 8][c0 + wv * 16 + lrow] = acc[r] * scale;
  }
  __syncthreads();

  // ---- row softmax (16 threads per row) ----
  {
    int row = tid >> 4, j = tid & 15;
    float mx = -3.0e38f;
    for (int c = j; c < 512; c += 16) mx = fmaxf(mx, S[row][c]);
    red[row][j] = mx;
    __syncthreads();
    float rm = red[row][0];
#pragma unroll
    for (int q2 = 1; q2 < 16; ++q2) rm = fmaxf(rm, red[row][q2]);
    __syncthreads();
    float sum = 0.0f;
    for (int c = j; c < 512; c += 16) {
      float e = __expf(S[row][c] - rm);
      S[row][c] = e;
      sum += e;
    }
    red[row][j] = sum;
    __syncthreads();
    float tt = 0.0f;
#pragma unroll
    for (int q2 = 0; q2 < 16; ++q2) tt += red[row][q2];
    float inv = 1.0f / tt;
    for (int c = j; c < 512; c += 16) S[row][c] *= inv;
  }
  __syncthreads();

  // ---- out = P @ V: waves split (dim quadrant x K half) ----
  int n_off = (wv & 3) * 16;
  int khalf = wv >> 2;
  v8f oacc = {};
  for (int t0 = 0; t0 < 512; t0 += 64) {
    {  // stage V chunk transposed: 64 tokens x 64 dims
      int tt = t0 + (tid >> 2), d0 = (tid & 3) * 16;
      const float* vp = qkv + ((size_t)(b * 512 + tt)) * 1536 + 1024 + h * 64 + d0;
#pragma unroll
      for (int i = 0; i < 16; ++i) Vt[d0 + i][tt - t0] = (__bf16)vp[i];
    }
    __syncthreads();
    int tk = khalf * 32;
    v16bf af, bfr;
#pragma unroll
    for (int j = 0; j < 8; ++j) {
      af[j]     = (__bf16)S[lrow][t0 + tk + lhi * 16 + j];
      af[8 + j] = (__bf16)S[lrow][t0 + tk + lhi * 16 + 8 + j];
    }
#pragma unroll
    for (int e = 0; e < 16; ++e) bfr[e] = Vt[n_off + lrow][tk + lhi * 16 + e];
    oacc = WMMA_BF16(af, bfr, oacc);
    __syncthreads();
  }
#pragma unroll
  for (int r = 0; r < 8; ++r) Oacc[khalf][r + lhi * 8][n_off + lrow] = oacc[r];
  __syncthreads();

  // ---- combine K halves, gate-mix with KNN memory output, write ----
  {
    int row = tid >> 4, d0 = (tid & 15) * 4;
    float g = layer0 ? (1.0f / (1.0f + __expf(-gate[h]))) : 0.0f;
#pragma unroll
    for (int i = 0; i < 4; ++i) {
      int d = d0 + i;
      float o = Oacc[0][row][d] + Oacc[1][row][d];
      if (layer0) {
        float mo = memout[((size_t)((b * 8 + h) * 512 + s0 + row)) * 64 + d];
        o = o * (1.0f - g) + mo * g;
      }
      attnout[((size_t)(b * 512 + s0 + row)) * 512 + h * 64 + d] = o;
    }
  }
}

// ---------------------------------------------------------------------------
// Kernel 4: KNN memory layer: sim = q @ mem_k^T (WMMA), per-row top-32
// (sorted insertion in LDS), softmax over 32, weighted gather of mem_v.
// One block per (b, h, 16-row q tile).
// ---------------------------------------------------------------------------
__global__ __launch_bounds__(256) void knn_kernel(
    const float* __restrict__ qkv, const float* __restrict__ mem_k,
    const float* __restrict__ mem_v, float* __restrict__ memout) {
  int bid = blockIdx.x;
  int st = bid & 31, h = (bid >> 5) & 7, b = bid >> 8;
  int s0 = st * 16;
  __shared__ __bf16 qs[16][72];
  __shared__ float  simb[16][132];
  __shared__ float  topv[16][32];
  __shared__ int    topi[16][32];
  __shared__ int    cnt[16];
  __shared__ float  prob[16][32];

  int tid = threadIdx.x, wv = tid >> 5, lane = tid & 31;
  int lrow = lane & 15, lhi = lane >> 4;
  const float scale = 0.125f;

  {
    int row = tid >> 4, f0 = (tid & 15) * 4;
    const float* qp = qkv + ((size_t)(b * 512 + s0 + row)) * 1536 + h * 64 + f0;
#pragma unroll
    for (int i = 0; i < 4; ++i) qs[row][f0 + i] = (__bf16)qp[i];
  }
  if (tid < 16) cnt[tid] = 0;
  __syncthreads();

  for (int m0 = 0; m0 < 8192; m0 += 128) {
    int mrow = m0 + wv * 16 + lrow;
    const float* kp = mem_k + ((size_t)b * 8192 + mrow) * 64;
    v8f acc = {};
#pragma unroll
    for (int ks = 0; ks < 2; ++ks) {
      v16bf af, bfr;
#pragma unroll
      for (int j = 0; j < 8; ++j) {
        af[j]     = qs[lrow][ks * 32 + lhi * 16 + j];
        af[8 + j] = qs[lrow][ks * 32 + lhi * 16 + 8 + j];
      }
#pragma unroll
      for (int e = 0; e < 16; ++e) bfr[e] = (__bf16)kp[ks * 32 + lhi * 16 + e];
      acc = WMMA_BF16(af, bfr, acc);
    }
#pragma unroll
    for (int r = 0; r < 8; ++r)
      simb[r + lhi * 8][wv * 16 + lrow] = acc[r] * scale;
    __syncthreads();

    if (tid < 16) {  // per-row sorted top-32 insertion
      int row = tid;
      int c = cnt[row];
      for (int j = 0; j < 128; ++j) {
        float v = simb[row][j];
        int idx = m0 + j;
        if (c < 32) {
          int p = c++;
          while (p > 0 && topv[row][p - 1] < v) {
            topv[row][p] = topv[row][p - 1];
            topi[row][p] = topi[row][p - 1];
            --p;
          }
          topv[row][p] = v;
          topi[row][p] = idx;
        } else if (v > topv[row][31]) {
          int p = 31;
          while (p > 0 && topv[row][p - 1] < v) {
            topv[row][p] = topv[row][p - 1];
            topi[row][p] = topi[row][p - 1];
            --p;
          }
          topv[row][p] = v;
          topi[row][p] = idx;
        }
      }
      cnt[row] = c;
    }
    __syncthreads();
  }

  if (tid < 16) {  // softmax over top-32 (sorted desc => max at [0])
    int row = tid;
    float mx = topv[row][0];
    float sm = 0.0f;
#pragma unroll
    for (int k = 0; k < 32; ++k) {
      float e = __expf(topv[row][k] - mx);
      prob[row][k] = e;
      sm += e;
    }
    float inv = 1.0f / sm;
#pragma unroll
    for (int k = 0; k < 32; ++k) prob[row][k] *= inv;
  }
  __syncthreads();

  {  // weighted gather of mem_v
    int row = tid >> 4, d0 = (tid & 15) * 4;
    float o0 = 0, o1 = 0, o2 = 0, o3 = 0;
#pragma unroll 4
    for (int k = 0; k < 32; ++k) {
      float p = prob[row][k];
      const float* vp = mem_v + ((size_t)b * 8192 + topi[row][k]) * 64 + d0;
      o0 += p * vp[0];
      o1 += p * vp[1];
      o2 += p * vp[2];
      o3 += p * vp[3];
    }
    float* op = memout + ((size_t)((b * 8 + h) * 512 + s0 + row)) * 64 + d0;
    op[0] = o0; op[1] = o1; op[2] = o2; op[3] = o3;
  }
}

// ---------------------------------------------------------------------------
// Kernel 5: tok = LayerNorm(tok + delta) in place, one block per row (D=512)
// ---------------------------------------------------------------------------
__global__ __launch_bounds__(256) void add_ln_kernel(
    float* __restrict__ tok, const float* __restrict__ delta,
    const float* __restrict__ g, const float* __restrict__ bta) {
  int row = blockIdx.x;
  int tid = threadIdx.x;
  __shared__ float red[256];
  float* tp = tok + (size_t)row * 512;
  const float* dp = delta + (size_t)row * 512;
  float v0 = tp[tid] + dp[tid];
  float v1 = tp[tid + 256] + dp[tid + 256];

  red[tid] = v0 + v1;
  __syncthreads();
  for (int s = 128; s > 0; s >>= 1) {
    if (tid < s) red[tid] += red[tid + s];
    __syncthreads();
  }
  float mean = red[0] * (1.0f / 512.0f);
  __syncthreads();
  float c0 = v0 - mean, c1 = v1 - mean;
  red[tid] = c0 * c0 + c1 * c1;
  __syncthreads();
  for (int s = 128; s > 0; s >>= 1) {
    if (tid < s) red[tid] += red[tid + s];
    __syncthreads();
  }
  float rstd = rsqrtf(red[0] * (1.0f / 512.0f) + 1e-5f);
  tp[tid]       = c0 * rstd * g[tid] + bta[tid];
  tp[tid + 256] = c1 * rstd * g[tid + 256] + bta[tid + 256];
}

// ---------------------------------------------------------------------------
// Kernel 6-8: pool + classifier head
// ---------------------------------------------------------------------------
__global__ __launch_bounds__(256) void pool_kernel(const float* __restrict__ tok,
                                                   float* __restrict__ pooled) {
  int i = blockIdx.x * 256 + threadIdx.x;  // 0..2047
  int b = i >> 9, d = i & 511;
  float s = 0.0f;
  for (int t = 0; t < 512; ++t) s += tok[((size_t)(b * 512 + t)) * 512 + d];
  pooled[i] = s * (1.0f / 512.0f);
}

__global__ __launch_bounds__(256) void head1_kernel(
    const float* __restrict__ pooled, const float* __restrict__ w,
    const float* __restrict__ bias, float* __restrict__ hbuf) {
  int i = blockIdx.x * 256 + threadIdx.x;  // 0..4095
  int b = i >> 10, j = i & 1023;
  float s = bias[j];
  for (int k = 0; k < 512; ++k) s += pooled[b * 512 + k] * w[(size_t)k * 1024 + j];
  hbuf[i] = gelu_exact(s);
}

__global__ __launch_bounds__(256) void head2_kernel(
    const float* __restrict__ hbuf, const float* __restrict__ w,
    const float* __restrict__ bias, float* __restrict__ out) {
  int i = blockIdx.x * 256 + threadIdx.x;
  if (i >= 4 * 117) return;
  int b = i / 117, c = i % 117;
  float s = bias[c];
  for (int k = 0; k < 1024; ++k) s += hbuf[b * 1024 + k] * w[(size_t)k * 117 + c];
  out[i] = s;
}

// ---------------------------------------------------------------------------
extern "C" void kernel_launch(void* const* d_in, const int* in_sizes, int n_in,
                              void* d_out, int out_size, void* d_ws, size_t ws_size,
                              hipStream_t stream) {
  const float* x     = (const float*)d_in[0];
  const float* mask  = (const float*)d_in[1];
  const float* pe    = (const float*)d_in[2];
  const float* Wqkv  = (const float*)d_in[3];
  const float* bqkv  = (const float*)d_in[4];
  const float* Wo    = (const float*)d_in[5];
  const float* bo    = (const float*)d_in[6];
  const float* ln1_g = (const float*)d_in[7];
  const float* ln1_b = (const float*)d_in[8];
  const float* W1    = (const float*)d_in[9];
  const float* b1    = (const float*)d_in[10];
  const float* W2    = (const float*)d_in[11];
  const float* b2    = (const float*)d_in[12];
  const float* ln2_g = (const float*)d_in[13];
  const float* ln2_b = (const float*)d_in[14];
  const float* gate  = (const float*)d_in[15];
  const float* mem_k = (const float*)d_in[16];
  const float* mem_v = (const float*)d_in[17];
  const float* fc1_w = (const float*)d_in[18];
  const float* fc1_b = (const float*)d_in[19];
  const float* fc2_w = (const float*)d_in[20];
  const float* fc2_b = (const float*)d_in[21];
  float* out = (float*)d_out;

  float* ws = (float*)d_ws;
  size_t o = 0;
  float* tok     = ws + o; o += (size_t)4 * 512 * 512;      // 1M
  float* qkv     = ws + o; o += (size_t)4 * 512 * 1536;     // 3M
  float* attnout = ws + o; o += (size_t)4 * 512 * 512;      // 1M
  float* memout  = ws + o; o += (size_t)4 * 512 * 512;      // 1M
  float* bigb    = ws + o; o += (size_t)2048 * 2048;        // 4M (FF hidden)
  float* smallb  = ws + o; o += (size_t)2048 * 512;         // 1M
  float* pooled  = ws + o; o += 2048;
  float* hbuf    = ws + o; o += 4096;

  // features * resized mask + positional encoding -> tokens [B,S,D]
  tok_init_kernel<<<2048, 256, 0, stream>>>(x, mask, pe, tok);

  for (int l = 0; l < 6; ++l) {
    // QKV projection: [2048,512] @ [512,1536]
    gemm_bf16_kernel<<<dim3(1536 / 64, 2048 / 128), 256, 0, stream>>>(
        tok, Wqkv + (size_t)l * 512 * 1536, bqkv + (size_t)l * 1536, qkv,
        2048, 1536, 512, 0);
    if (l == 0)
      knn_kernel<<<1024, 256, 0, stream>>>(qkv, mem_k, mem_v, memout);
    attn_kernel<<<1024, 256, 0, stream>>>(qkv, memout, gate, attnout, l == 0 ? 1 : 0);
    // output projection
    gemm_bf16_kernel<<<dim3(512 / 64, 2048 / 128), 256, 0, stream>>>(
        attnout, Wo + (size_t)l * 512 * 512, bo + (size_t)l * 512, smallb,
        2048, 512, 512, 0);
    add_ln_kernel<<<2048, 256, 0, stream>>>(tok, smallb, ln1_g + (size_t)l * 512,
                                            ln1_b + (size_t)l * 512);
    // FFN
    gemm_bf16_kernel<<<dim3(2048 / 64, 2048 / 128), 256, 0, stream>>>(
        tok, W1 + (size_t)l * 512 * 2048, b1 + (size_t)l * 2048, bigb,
        2048, 2048, 512, 1);
    gemm_bf16_kernel<<<dim3(512 / 64, 2048 / 128), 256, 0, stream>>>(
        bigb, W2 + (size_t)l * 2048 * 512, b2 + (size_t)l * 512, smallb,
        2048, 512, 2048, 0);
    add_ln_kernel<<<2048, 256, 0, stream>>>(tok, smallb, ln2_g + (size_t)l * 512,
                                            ln2_b + (size_t)l * 512);
  }

  pool_kernel<<<8, 256, 0, stream>>>(tok, pooled);
  head1_kernel<<<16, 256, 0, stream>>>(pooled, fc1_w, fc1_b, hbuf);
  head2_kernel<<<2, 256, 0, stream>>>(hbuf, fc2_w, fc2_b, out);
}
